// SPECTRELayer_21869973471531
// MI455X (gfx1250) — compile-verified, matrix operand order checked by reading, not run on hardware
//
#include <hip/hip_runtime.h>
#include <math.h>

// ---------------------------------------------------------------------------
// SPECTRE layer for gfx1250 (MI455X). B=4, L=4096, D=1024, H=16, DH=64.
// ---------------------------------------------------------------------------
#define B_  4
#define L_  4096
#define D_  1024
#define H_  16
#define DH_ 64
#define NF_MAX_ 4097   // MAX_SEQ/2+1
#define NF_ 2049       // L/2+1

typedef __attribute__((ext_vector_type(16))) __bf16 v16bf;
typedef __attribute__((ext_vector_type(8)))  float  v8f;
typedef __attribute__((ext_vector_type(4)))  int    v4i;

// gfx1250 async global->LDS path (ASYNCcnt-tracked), guarded so the file
// still compiles on toolchains without the builtins.
#if defined(__has_builtin)
#if __has_builtin(__builtin_amdgcn_global_load_async_to_lds_b128) && \
    __has_builtin(__builtin_amdgcn_s_wait_asynccnt)
#define HAVE_ASYNC_LDS 1
#endif
#endif
#ifndef HAVE_ASYNC_LDS
#define HAVE_ASYNC_LDS 0
#endif

#if HAVE_ASYNC_LDS
typedef __attribute__((address_space(1))) v4i as1_v4i;
typedef __attribute__((address_space(3))) v4i as3_v4i;
#endif

__device__ __forceinline__ void copy16_g2l(const __bf16* g, __bf16* l) {
#if HAVE_ASYNC_LDS
  __builtin_amdgcn_global_load_async_to_lds_b128((as1_v4i*)g, (as3_v4i*)l,
                                                 0, 0);
#else
  *reinterpret_cast<uint4*>(l) = *reinterpret_cast<const uint4*>(g);
#endif
}

__device__ __forceinline__ void wait_async_lds() {
#if HAVE_ASYNC_LDS
  __builtin_amdgcn_s_wait_asynccnt(0);
#endif
}

// ---------------------------------------------------------------------------
// fp32 -> bf16 elementwise convert (grid-stride)
// ---------------------------------------------------------------------------
__global__ __launch_bounds__(256) void k_convert_bf16(const float* __restrict__ in,
                                                      __bf16* __restrict__ out,
                                                      long n) {
  long i = (long)blockIdx.x * blockDim.x + threadIdx.x;
  long stride = (long)gridDim.x * blockDim.x;
  for (; i < n; i += stride) out[i] = (__bf16)in[i];
}

// ---------------------------------------------------------------------------
// xbar[b,d] = mean over L of x[b,l,d]
// ---------------------------------------------------------------------------
__global__ __launch_bounds__(256) void k_xbar(const float* __restrict__ x,
                                              float* __restrict__ xbar) {
  int g = blockIdx.x * blockDim.x + threadIdx.x;    // over B_*D_
  if (g >= B_ * D_) return;
  int b = g >> 10;          // /D_
  int d = g & (D_ - 1);
  const float* p = x + (size_t)b * L_ * D_ + d;
  float s = 0.f;
  for (int l = 0; l < L_; ++l) s += p[(size_t)l * D_];
  xbar[g] = s * (1.0f / (float)L_);
}

// ---------------------------------------------------------------------------
// Per (b,h) gate/gw computation (q_bar = xbar @ Wq^T: mean commutes with GEMM)
// One block per (b,h). 64 blocks x 256 threads.
// ---------------------------------------------------------------------------
__global__ __launch_bounds__(256) void k_gate(
    const float* __restrict__ xbar, const float* __restrict__ Wq,
    const float* __restrict__ ln_g_s, const float* __restrict__ ln_g_b,
    const float* __restrict__ W1, const float* __restrict__ b1,
    const float* __restrict__ W2, const float* __restrict__ b2,
    const float* __restrict__ ln_w_s, const float* __restrict__ ln_w_b,
    const float* __restrict__ Ww1, const float* __restrict__ bw1,
    const float* __restrict__ Ww2, const float* __restrict__ bw2,
    float* __restrict__ gate_re, float* __restrict__ gate_im,
    float* __restrict__ gw1p) {
  __shared__ float xb[D_];
  __shared__ float qb[DH_];
  __shared__ float lng[DH_];
  __shared__ float lnw[DH_];
  __shared__ float h1[4 * DH_];
  __shared__ float a1[DH_];
  __shared__ float stats[2];

  int bh = blockIdx.x;            // b*H + h
  int b  = bh >> 4;
  int h  = bh & (H_ - 1);
  int tid = threadIdx.x;

  for (int i = tid; i < D_; i += 256) xb[i] = xbar[b * D_ + i];
  __syncthreads();

  if (tid < DH_) {
    const float* wrow = Wq + (size_t)(h * DH_ + tid) * D_;
    float s = 0.f;
    for (int d = 0; d < D_; ++d) s += xb[d] * wrow[d];
    qb[tid] = s;
  }
  __syncthreads();

  if (tid == 0) {
    float m = 0.f;
    for (int i = 0; i < DH_; ++i) m += qb[i];
    m *= (1.0f / DH_);
    float v = 0.f;
    for (int i = 0; i < DH_; ++i) { float d = qb[i] - m; v += d * d; }
    v *= (1.0f / DH_);
    stats[0] = m;
    stats[1] = rsqrtf(v + 1e-5f);
  }
  __syncthreads();

  if (tid < DH_) {
    float t = (qb[tid] - stats[0]) * stats[1];
    lng[tid] = t * ln_g_s[tid] + ln_g_b[tid];
    lnw[tid] = t * ln_w_s[tid] + ln_w_b[tid];
  }
  __syncthreads();

  // h1 = gelu(lng @ W1^T + b1), 256 outputs, one per thread
  {
    const float* wrow = W1 + (size_t)tid * DH_;
    float s = b1[tid];
    for (int d = 0; d < DH_; ++d) s += lng[d] * wrow[d];
    h1[tid] = 0.5f * s * (1.f + erff(s * 0.70710678118654752f));
  }
  // a1 = gelu(lnw @ Ww1^T + bw1)
  if (tid < DH_) {
    const float* wrow = Ww1 + (size_t)tid * DH_;
    float s = bw1[tid];
    for (int d = 0; d < DH_; ++d) s += lnw[d] * wrow[d];
    a1[tid] = 0.5f * s * (1.f + erff(s * 0.70710678118654752f));
  }
  __syncthreads();

  if (tid < DH_) {
    const float* wrow = Ww2 + (size_t)tid * DH_;
    float s = bw2[tid];
    for (int d = 0; d < DH_; ++d) s += a1[d] * wrow[d];
    gw1p[bh * DH_ + tid] = 1.f + 1.f / (1.f + expf(-s));
  }

  // complex gates: g[:,:n_freq] + i*g[:,NF_MAX:NF_MAX+n_freq]
  for (int idx = tid; idx < 2 * NF_; idx += 256) {
    int part = (idx >= NF_) ? 1 : 0;
    int f = part ? (idx - NF_) : idx;
    int row = part ? (NF_MAX_ + f) : f;
    const float* wrow = W2 + (size_t)row * (4 * DH_);
    float s = b2[row];
    for (int d = 0; d < 4 * DH_; ++d) s += h1[d] * wrow[d];
    if (part) gate_im[(size_t)bh * NF_ + f] = s;
    else      gate_re[(size_t)bh * NF_ + f] = s;
  }
}

// ---------------------------------------------------------------------------
// bf16 WMMA GEMM: out = A(Mtot x K) * Bw(Ntot x K)^T, f32 accumulate.
// Block tile 128x64, 8 wave32 waves in a 4x2 grid, each wave: 2x2 of 16x16.
// Double-buffered LDS tiles fed by async global->LDS loads (ASYNCcnt):
//   wait(asynccnt 0); barrier; issue tile k+1 (other buffer); WMMA on tile k.
// OUT_MODE 0: out[m*Ntot + n]              (plain row-major, (B,L,D))
// OUT_MODE 1: out[((b*H+h)*DH+dh)*L + l]   (v in (B,H,DH,L) for the FFT)
// ---------------------------------------------------------------------------
template <int OUT_MODE>
__global__ __launch_bounds__(256) void k_gemm_wmma(
    const __bf16* __restrict__ A, const __bf16* __restrict__ Bw,
    float* __restrict__ out, int Mtot, int K, int Ntot) {
  constexpr int BM = 128, BN = 64, BK = 32, PAD = 8;
  __shared__ __align__(16) __bf16 As[2][BM][BK + PAD];
  __shared__ __align__(16) __bf16 Bs[2][BN][BK + PAD];

  const int tid  = threadIdx.x;
  const int wave = tid >> 5;
  const int lane = tid & 31;
  const int wm = wave & 3;          // 4 waves along M
  const int wn = wave >> 2;         // 2 waves along N
  const int m0 = blockIdx.x * BM;
  const int n0 = blockIdx.y * BN;
  const int lr = lane & 15;         // row (A) / col (B) within 16
  const int kg = lane >> 4;         // K-half group select

  // Per-thread staging assignment: 2 A-chunks + 1 B-chunk of 16B each.
  const int rowA0 = tid >> 2;             // 0..63
  const int rowA1 = rowA0 + 64;           // 64..127
  const int colA  = (tid & 3) * 8;
  const int rowB  = tid >> 2;
  const int colB  = (tid & 3) * 8;
  const __bf16* gA0 = A + (size_t)(m0 + rowA0) * K + colA;
  const __bf16* gA1 = A + (size_t)(m0 + rowA1) * K + colA;
  const __bf16* gB  = Bw + (size_t)(n0 + rowB) * K + colB;

  const v8f vzero = {0.f, 0.f, 0.f, 0.f, 0.f, 0.f, 0.f, 0.f};
  v8f acc[2][2];
  acc[0][0] = vzero; acc[0][1] = vzero; acc[1][0] = vzero; acc[1][1] = vzero;

  union Frag { uint4 q[2]; v16bf v; };

  auto stage = [&](int k0, int buf) {
    copy16_g2l(gA0 + k0, &As[buf][rowA0][colA]);
    copy16_g2l(gA1 + k0, &As[buf][rowA1][colA]);
    copy16_g2l(gB + k0, &Bs[buf][rowB][colB]);
  };

  stage(0, 0);
  const int nk = K / BK;
  for (int i = 0; i < nk; ++i) {
    const int cur = i & 1;
    wait_async_lds();
    __syncthreads();
    if (i + 1 < nk) stage((i + 1) * BK, cur ^ 1);

    // fragments per ISA 16-bit A/B layout: lane holds K kg*8..kg*8+7 and
    // 16+kg*8..16+kg*8+7 of its row/column.
    Frag af[2], bf2[2];
    for (int mt = 0; mt < 2; ++mt) {
      int r = wm * 32 + mt * 16 + lr;
      af[mt].q[0] = *reinterpret_cast<const uint4*>(&As[cur][r][kg * 8]);
      af[mt].q[1] = *reinterpret_cast<const uint4*>(&As[cur][r][16 + kg * 8]);
    }
    for (int nt = 0; nt < 2; ++nt) {
      int r = wn * 32 + nt * 16 + lr;
      bf2[nt].q[0] = *reinterpret_cast<const uint4*>(&Bs[cur][r][kg * 8]);
      bf2[nt].q[1] = *reinterpret_cast<const uint4*>(&Bs[cur][r][16 + kg * 8]);
    }

    for (int mt = 0; mt < 2; ++mt)
      for (int nt = 0; nt < 2; ++nt)
        acc[mt][nt] = __builtin_amdgcn_wmma_f32_16x16x32_bf16(
            false, af[mt].v, false, bf2[nt].v, (short)0, acc[mt][nt],
            false, false);
  }

  // C/D layout: VGPR r -> M = r + 8*kg, N = lane&15
  for (int mt = 0; mt < 2; ++mt) {
    for (int nt = 0; nt < 2; ++nt) {
      for (int r = 0; r < 8; ++r) {
        int gm = m0 + wm * 32 + mt * 16 + kg * 8 + r;
        int gn = n0 + wn * 32 + nt * 16 + lr;
        float val = acc[mt][nt][r];
        if (OUT_MODE == 0) {
          out[(size_t)gm * Ntot + gn] = val;
        } else {
          int b  = gm >> 12;          // /L_
          int l  = gm & (L_ - 1);
          int h  = gn >> 6;           // /DH_
          int dh = gn & (DH_ - 1);
          out[(((size_t)(b * H_ + h)) * DH_ + dh) * L_ + l] = val;
        }
      }
    }
  }
}

// ---------------------------------------------------------------------------
// In-LDS radix-2 FFT over 4096 points, 256 threads.  sign=-1 fwd, +1 inv.
// ---------------------------------------------------------------------------
__device__ inline void fft4096(float* re, float* im, float sign) {
  const int tid = threadIdx.x;
  // bit-reverse permutation (12 bits)
  for (int i = tid; i < L_; i += 256) {
    int j = (int)(__brev((unsigned)i) >> 20);
    if (i < j) {
      float tr = re[i]; re[i] = re[j]; re[j] = tr;
      float ti = im[i]; im[i] = im[j]; im[j] = ti;
    }
  }
  __syncthreads();
  for (int s = 1; s <= 12; ++s) {
    const int m = 1 << s;
    const int half = m >> 1;
    for (int k = tid; k < L_ / 2; k += 256) {
      int g  = k >> (s - 1);
      int j  = k & (half - 1);
      int i0 = (g << s) + j;
      int i1 = i0 + half;
      float ang = sign * 6.28318530717958647692f * (float)j / (float)m;
      float sw, cw;
      __sincosf(ang, &sw, &cw);
      float xr = re[i1], xi = im[i1];
      float tr = xr * cw - xi * sw;
      float ti = xr * sw + xi * cw;
      float ur = re[i0], ui = im[i0];
      re[i1] = ur - tr; im[i1] = ui - ti;
      re[i0] = ur + tr; im[i0] = ui + ti;
    }
    __syncthreads();
  }
}

// ---------------------------------------------------------------------------
// Fused spectral stage, one block per sequence (b,h,dh); in-place on vy.
// rfft -> gate -> modReLU -> Hermitian extend -> irfft -> *(1+gw).
// (The reference's wavelet branch algebraically reduces to y *= (1+gw).)
// ---------------------------------------------------------------------------
__global__ __launch_bounds__(256) void k_spectral(
    float* __restrict__ vy, const float* __restrict__ gate_re,
    const float* __restrict__ gate_im, const float* __restrict__ modrelu_bias,
    const float* __restrict__ gw1p) {
  __shared__ float re[L_];
  __shared__ float im[L_];
  const int seq = blockIdx.x;        // (b*H+h)*DH + dh
  const int dh  = seq & (DH_ - 1);
  const int bh  = seq >> 6;
  const int h   = bh & (H_ - 1);
  const int tid = threadIdx.x;
  const size_t base = (size_t)seq * L_;

  for (int i = tid; i < L_; i += 256) { re[i] = vy[base + i]; im[i] = 0.f; }
  __syncthreads();

  fft4096(re, im, -1.f);

  const float* gre = gate_re + (size_t)bh * NF_;
  const float* gim = gate_im + (size_t)bh * NF_;
  const float* mb  = modrelu_bias + (size_t)h * NF_MAX_;
  for (int f = tid; f < NF_; f += 256) {
    float xr = re[f], xi = im[f];
    float gr = gre[f], gi = gim[f];
    float yr = xr * gr - xi * gi;
    float yi = xr * gi + xi * gr;
    float mag = sqrtf(yr * yr + yi * yi);
    float sc = fmaxf(mag + mb[f], 0.f) / (mag + 1e-6f);
    re[f] = yr * sc;
    im[f] = yi * sc;
  }
  __syncthreads();
  // Hermitian symmetry for the full inverse transform
  for (int f = tid + NF_; f < L_; f += 256) {
    re[f] =  re[L_ - f];
    im[f] = -im[L_ - f];
  }
  __syncthreads();

  fft4096(re, im, +1.f);

  const float scale = (1.0f / (float)L_) * gw1p[bh * DH_ + dh];
  for (int i = tid; i < L_; i += 256) vy[base + i] = re[i] * scale;
}

// ---------------------------------------------------------------------------
// Transpose y (B,H,DH,L) f32 -> ybf (B,L,D) bf16 via 32x32 LDS tiles.
// grid: (L/32, DH/32, B*H), 256 threads (32x8).
// ---------------------------------------------------------------------------
__global__ __launch_bounds__(256) void k_transpose_bf16(
    const float* __restrict__ y, __bf16* __restrict__ out) {
  __shared__ float tile[32][33];
  const int bh = blockIdx.z;
  const int b  = bh >> 4;
  const int h  = bh & (H_ - 1);
  const int l0 = blockIdx.x * 32;
  const int d0 = blockIdx.y * 32;
  const int tx = threadIdx.x & 31;
  const int ty = threadIdx.x >> 5;

  for (int r = ty; r < 32; r += 8)
    tile[r][tx] = y[((size_t)bh * DH_ + d0 + r) * L_ + l0 + tx];
  __syncthreads();
  for (int r = ty; r < 32; r += 8)
    out[((size_t)b * L_ + l0 + r) * D_ + h * DH_ + d0 + tx] =
        (__bf16)tile[tx][r];
}

// ---------------------------------------------------------------------------
// launcher
// ---------------------------------------------------------------------------
extern "C" void kernel_launch(void* const* d_in, const int* in_sizes, int n_in,
                              void* d_out, int out_size, void* d_ws,
                              size_t ws_size, hipStream_t stream) {
  const float* x            = (const float*)d_in[0];
  const float* Wq           = (const float*)d_in[1];
  const float* Wv           = (const float*)d_in[2];
  const float* Wo           = (const float*)d_in[3];
  const float* ln_g_s       = (const float*)d_in[4];
  const float* ln_g_b       = (const float*)d_in[5];
  const float* W1           = (const float*)d_in[6];
  const float* b1           = (const float*)d_in[7];
  const float* W2           = (const float*)d_in[8];
  const float* b2           = (const float*)d_in[9];
  const float* modrelu_bias = (const float*)d_in[10];
  const float* ln_w_s       = (const float*)d_in[11];
  const float* ln_w_b       = (const float*)d_in[12];
  const float* Ww1          = (const float*)d_in[13];
  const float* bw1          = (const float*)d_in[14];
  const float* Ww2          = (const float*)d_in[15];
  const float* bw2          = (const float*)d_in[16];

  char* w = (char*)d_ws;
  const size_t XBF_BYTES = (size_t)B_ * L_ * D_ * 2;  // 32 MB (reused as ybf)
  const size_t WBF_BYTES = (size_t)D_ * D_ * 2;       // 2 MB
  __bf16* xbf  = (__bf16*)w;                          // also ybf later
  __bf16* Wvbf = (__bf16*)(w + XBF_BYTES);
  __bf16* Wobf = (__bf16*)(w + XBF_BYTES + WBF_BYTES);
  float*  vy   = (float*)(w + XBF_BYTES + 2 * WBF_BYTES);  // 64 MB, v then y
  char*   w2   = w + XBF_BYTES + 2 * WBF_BYTES + (size_t)B_ * L_ * D_ * 4;
  float* xbar    = (float*)w2;                // B*D
  float* gate_re = xbar + B_ * D_;            // B*H*NF_
  float* gate_im = gate_re + B_ * H_ * NF_;
  float* gw1p    = gate_im + B_ * H_ * NF_;   // B*H*DH

  const int Mtot = B_ * L_;

  // 1) bf16 staging of x, Wv, Wo
  k_convert_bf16<<<2048, 256, 0, stream>>>(x, xbf, (long)B_ * L_ * D_);
  k_convert_bf16<<<512, 256, 0, stream>>>(Wv, Wvbf, (long)D_ * D_);
  k_convert_bf16<<<512, 256, 0, stream>>>(Wo, Wobf, (long)D_ * D_);

  // 2) xbar = mean_L(x)   (q_bar = xbar @ Wq^T since mean commutes with GEMM)
  k_xbar<<<(B_ * D_ + 255) / 256, 256, 0, stream>>>(x, xbar);

  // 3) gates + (1+gw)
  k_gate<<<B_ * H_, 256, 0, stream>>>(xbar, Wq, ln_g_s, ln_g_b, W1, b1, W2, b2,
                                      ln_w_s, ln_w_b, Ww1, bw1, Ww2, bw2,
                                      gate_re, gate_im, gw1p);

  // 4) v = x @ Wv^T  (WMMA), written in (B,H,DH,L) for contiguous FFT reads
  {
    dim3 grid(Mtot / 128, D_ / 64);
    k_gemm_wmma<1><<<grid, 256, 0, stream>>>(xbf, Wvbf, vy, Mtot, D_, D_);
  }

  // 5) fused rfft/gate/modReLU/irfft/(1+gw) per sequence, in place
  k_spectral<<<B_ * H_ * DH_, 256, 0, stream>>>(vy, gate_re, gate_im,
                                                modrelu_bias, gw1p);

  // 6) y (B,H,DH,L) -> bf16 (B,L,D), reusing the xbf buffer
  {
    dim3 grid(L_ / 32, DH_ / 32, B_ * H_);
    k_transpose_bf16<<<grid, 256, 0, stream>>>(vy, xbf);
  }

  // 7) out = y @ Wo^T (WMMA)
  {
    dim3 grid(Mtot / 128, D_ / 64);
    k_gemm_wmma<0><<<grid, 256, 0, stream>>>(xbf, Wobf, (float*)d_out, Mtot,
                                             D_, D_);
  }
}
